// comb_encoder_4483945857038
// MI455X (gfx1250) — matrix operand order repaired and checked by reading, not compiled, not run on hardware
//
#include <hip/hip_runtime.h>
#include <hip/hip_bf16.h>
#include <stdint.h>

typedef __attribute__((ext_vector_type(16))) _Float16 v16h;
typedef __attribute__((ext_vector_type(8)))  _Float16 v8h;
typedef __attribute__((ext_vector_type(8)))  float    v8f;

#define T_LEN   4096
#define DIM     1024
#define HDIM    1024
#define G4      4096            // 4*H gate rows
#define NSEG    512
#define NWG_DIR 64              // workgroups per direction in recurrence
#define ROWS_WG 64              // gate rows per WG (16 h-idx * 4 gates)

// ---------------- elementwise f32 -> f16 ----------------
__global__ void f32_to_f16_kernel(const float* __restrict__ src,
                                  _Float16* __restrict__ dst, long n) {
  long i = (long)blockIdx.x * blockDim.x + threadIdx.x;
  long stride = (long)gridDim.x * blockDim.x;
  for (; i < n; i += stride) dst[i] = (_Float16)src[i];
}

// ---------------- WMMA GEMM: pre = A @ W^T + (bih + bhh) ----------------
// A: [T_LEN, K] f16 row-major; W: [2, G4, K] f16 row-major; pre: [2, T_LEN, G4] f32
// Each wave computes a 16(M) x 128(N) strip: 8 accumulator tiles, A fragment reused.
__global__ __launch_bounds__(256) void gemm_pre_kernel(
    const _Float16* __restrict__ A,
    const _Float16* __restrict__ W,
    const float* __restrict__ bih,      // [2, G4] layer slice
    const float* __restrict__ bhh,      // [2, G4] layer slice
    float* __restrict__ pre,
    int K)
{
  const int lane = threadIdx.x & 31;
  const int wv   = threadIdx.x >> 5;
  const int half = lane >> 4;
  const int col  = lane & 15;
  const int d    = blockIdx.z;
  const int m0   = blockIdx.y * 128 + wv * 16;
  const int n0   = blockIdx.x * 128;

  const _Float16* Wd   = W + (size_t)d * G4 * K;
  const _Float16* arow = A + (size_t)(m0 + col) * K;

  v8f acc[8] = {};

  for (int k0 = 0; k0 < K; k0 += 32) {
    // A fragment (16x32 f16): lane half h -> K = h*16 + e (contiguous)
    v16h a;
    {
      a = *(const v16h*)(arow + k0 + half * 16);
    }
#pragma unroll
    for (int j = 0; j < 8; ++j) {
      // B fragment (32x16 f16, from W^T): lane n=col, K = half*16 + e (contiguous 32B)
      const _Float16* wrow = Wd + (size_t)(n0 + j * 16 + col) * K + k0 + half * 16;
      v16h b = *(const v16h*)wrow;
      acc[j] = __builtin_amdgcn_wmma_f32_16x16x32_f16(
          false, a, false, b, (short)0, acc[j], false, false);
    }
  }

#pragma unroll
  for (int j = 0; j < 8; ++j) {
    const int n = n0 + j * 16 + col;
    const float bias = bih[d * G4 + n] + bhh[d * G4 + n];
#pragma unroll
    for (int v = 0; v < 8; ++v) {
      const int row = m0 + v + 8 * half;   // C/D layout: VGPR v, lane-half selects M
      pre[((size_t)d * T_LEN + row) * G4 + n] = acc[j][v] + bias;
    }
  }
}

// ---------------- persistent LSTM recurrence (both directions) ----------------
// pre: [2, T, G4] f32; whh: [2, G4, HDIM] f16 (layer slice); hsF/hsB: [T, H] f32
// Each WG owns 16 h-indices (64 gate rows) with its Whh slice resident in LDS.
__global__ __launch_bounds__(256) void lstm_recur_kernel(
    const float* __restrict__ pre,
    const _Float16* __restrict__ whh,
    float* __restrict__ hsF,
    float* __restrict__ hsB,
    float* __restrict__ cfin,           // [2, H]
    unsigned* __restrict__ counters)    // [2]
{
  extern __shared__ char smem[];
  _Float16* wlds = (_Float16*)smem;                                     // 64*1024 f16
  float*    hld  = (float*)(smem + (size_t)ROWS_WG * HDIM * 2);         // 1024 f32
  float*    gv   = hld + HDIM;                                          // 64 f32

  const int tid = threadIdx.x;
  const int d   = blockIdx.x / NWG_DIR;
  const int wg  = blockIdx.x % NWG_DIR;
  const int j0  = wg * 16;

  // preload 64 gate rows of Whh into LDS (u64 copies)
  const _Float16* wsrc = whh + (size_t)d * G4 * HDIM;
  for (int i = tid; i < ROWS_WG * (HDIM / 4); i += 256) {
    int lr = i / (HDIM / 4);
    int q  = i % (HDIM / 4);
    int grow = (lr >> 4) * HDIM + j0 + (lr & 15);
    ((unsigned long long*)wlds)[(size_t)lr * (HDIM / 4) + q] =
        ((const unsigned long long*)(wsrc + (size_t)grow * HDIM))[q];
  }
  __syncthreads();

  const int lr   = tid >> 2;                 // 0..63 local gate row
  const int part = tid & 3;                  // K quarter
  const int grow = (lr >> 4) * HDIM + j0 + (lr & 15);

  float c_state = 0.0f;                      // valid for tid < 16
  float* hs = (d == 0) ? hsF : hsB;

  for (int s = 0; s < T_LEN; ++s) {
    const int tr = (d == 0) ? s : (T_LEN - 1 - s);

    // stage h_prev into LDS
    if (s == 0) {
      for (int i = tid; i < HDIM; i += 256) hld[i] = 0.0f;
    } else {
      const float* hprev = hs + (size_t)(s - 1) * HDIM;
      for (int i = tid; i < HDIM; i += 256) hld[i] = hprev[i];
    }
    __syncthreads();

    // 256-wide partial dot product (f16 weights from LDS)
    const v8h*    wp = (const v8h*)(wlds + (size_t)lr * HDIM + part * 256);
    const float4* hp = (const float4*)(hld + part * 256);
    float sum = 0.0f;
#pragma unroll 4
    for (int q = 0; q < 32; ++q) {
      v8h w8 = wp[q];
      float4 ha = hp[2 * q], hb = hp[2 * q + 1];
      sum += (float)w8[0] * ha.x + (float)w8[1] * ha.y +
             (float)w8[2] * ha.z + (float)w8[3] * ha.w +
             (float)w8[4] * hb.x + (float)w8[5] * hb.y +
             (float)w8[6] * hb.z + (float)w8[7] * hb.w;
    }
    sum += __shfl_down(sum, 1, 4);
    sum += __shfl_down(sum, 2, 4);
    if (part == 0)
      gv[lr] = sum + pre[((size_t)d * T_LEN + tr) * G4 + grow];
    __syncthreads();

    if (tid < 16) {
      float gi = gv[tid];
      float gf = gv[16 + tid];
      float gg = gv[32 + tid];
      float go = gv[48 + tid];
      float ig = 1.0f / (1.0f + __expf(-gi));
      float fg = 1.0f / (1.0f + __expf(-gf));
      float g  = tanhf(gg);
      float og = 1.0f / (1.0f + __expf(-go));
      c_state = fg * c_state + ig * g;
      float h = og * tanhf(c_state);
      hs[(size_t)s * HDIM + j0 + tid] = h;
    }

    // per-direction grid barrier
    __threadfence();
    __syncthreads();
    if (tid == 0) {
      __hip_atomic_fetch_add(counters + d, 1u, __ATOMIC_RELEASE,
                             __HIP_MEMORY_SCOPE_AGENT);
      unsigned target = (unsigned)(s + 1) * NWG_DIR;
      while (__hip_atomic_load(counters + d, __ATOMIC_ACQUIRE,
                               __HIP_MEMORY_SCOPE_AGENT) < target)
        __builtin_amdgcn_s_sleep(1);
    }
    __syncthreads();
  }

  if (tid < 16) cfin[(size_t)d * HDIM + j0 + tid] = c_state;
}

// ---------------- build layer-1 input: concat(hf, reversed hb) -> f16 ----------------
__global__ void build_x1_kernel(const float* __restrict__ hf,
                                const float* __restrict__ hb,
                                _Float16* __restrict__ x1) {
  long i = (long)blockIdx.x * 256 + threadIdx.x;
  if (i >= (long)T_LEN * 2048) return;
  int t = (int)(i >> 11);
  int j = (int)(i & 2047);
  float v = (j < HDIM) ? hf[(size_t)t * HDIM + j]
                       : hb[(size_t)(T_LEN - 1 - t) * HDIM + (j - HDIM)];
  x1[i] = (_Float16)v;
}

// ---------------- output_t = concat(hf1, reversed hb1) ----------------
__global__ void finalize_out_kernel(const float* __restrict__ hf,
                                    const float* __restrict__ hb,
                                    float* __restrict__ outp) {
  long i = (long)blockIdx.x * 256 + threadIdx.x;
  if (i >= (long)T_LEN * 2048) return;
  int t = (int)(i >> 11);
  int j = (int)(i & 2047);
  outp[i] = (j < HDIM) ? hf[(size_t)t * HDIM + j]
                       : hb[(size_t)(T_LEN - 1 - t) * HDIM + (j - HDIM)];
}

// ---------------- h_n / c_n ----------------
__global__ void finalize_states_kernel(const float* hf0, const float* hb0,
                                       const float* hf1, const float* hb1,
                                       const float* cfin, float* hn, float* cn) {
  int i = blockIdx.x * 256 + threadIdx.x;   // 0..8191
  if (i < 4096) {
    int k = i >> 10, j = i & 1023;
    const float* src = (k == 0) ? hf0 : (k == 1) ? hb0 : (k == 2) ? hf1 : hb1;
    hn[i] = src[(size_t)(T_LEN - 1) * HDIM + j];
  } else if (i < 8192) {
    cn[i - 4096] = cfin[i - 4096];
  }
}

// ---------------- segment mean pooling (sorted ids, binary search) ----------------
__global__ __launch_bounds__(256) void segpool_kernel(
    const float* __restrict__ outp,      // [T, 2048]
    const int* __restrict__ seg,         // [T] sorted
    float* __restrict__ rep)             // [NSEG-2, 2048]
{
  int s = blockIdx.x;
  int lo = 0, hi = T_LEN;
  while (lo < hi) { int mid = (lo + hi) >> 1; if (seg[mid] < s) lo = mid + 1; else hi = mid; }
  int start = lo;
  lo = start; hi = T_LEN;
  while (lo < hi) { int mid = (lo + hi) >> 1; if (seg[mid] < s + 1) lo = mid + 1; else hi = mid; }
  int end = lo;
  float inv = 1.0f / ((float)(end - start) + 1.0f);
  if (s == 0 || s == NSEG - 1) return;
  for (int j = threadIdx.x; j < 2048; j += 256) {
    float acc = 0.0f;
    for (int r = start; r < end; ++r) acc += outp[(size_t)r * 2048 + j];
    rep[(size_t)(s - 1) * 2048 + j] = acc * inv;
  }
}

extern "C" void kernel_launch(void* const* d_in, const int* in_sizes, int n_in,
                              void* d_out, int out_size, void* d_ws, size_t ws_size,
                              hipStream_t stream) {
  (void)in_sizes; (void)n_in; (void)out_size; (void)ws_size;

  const float* input_t     = (const float*)d_in[0];
  const int*   segment_ids = (const int*)d_in[1];
  const float* W_ih0       = (const float*)d_in[3];
  const float* W_ih1       = (const float*)d_in[4];
  const float* W_hh        = (const float*)d_in[5];
  const float* b_ih        = (const float*)d_in[6];
  const float* b_hh        = (const float*)d_in[7];

  char* ws = (char*)d_ws;
  size_t off = 0;
  auto alloc = [&](size_t bytes) -> void* {
    void* p = ws + off;
    off = (off + bytes + 255) & ~(size_t)255;
    return p;
  };
  _Float16* x16    = (_Float16*)alloc((size_t)T_LEN * 2048 * 2);      // 16 MB
  _Float16* wih16  = (_Float16*)alloc((size_t)2 * G4 * 2048 * 2);     // 32 MB
  _Float16* whh16  = (_Float16*)alloc((size_t)4 * G4 * HDIM * 2);     // 32 MB
  float*    pre    = (float*)alloc((size_t)2 * T_LEN * G4 * 4);       // 128 MB
  float*    hf0    = (float*)alloc((size_t)T_LEN * HDIM * 4);
  float*    hb0    = (float*)alloc((size_t)T_LEN * HDIM * 4);
  float*    hf1    = (float*)alloc((size_t)T_LEN * HDIM * 4);
  float*    hb1    = (float*)alloc((size_t)T_LEN * HDIM * 4);
  float*    cfin   = (float*)alloc((size_t)4 * HDIM * 4);
  unsigned* counters = (unsigned*)alloc(256);

  hipMemsetAsync(counters, 0, 256, stream);

  // f16 conversions
  f32_to_f16_kernel<<<2048, 256, 0, stream>>>(input_t, x16, (long)T_LEN * DIM);
  f32_to_f16_kernel<<<4096, 256, 0, stream>>>(W_hh, whh16, (long)4 * G4 * HDIM);
  f32_to_f16_kernel<<<4096, 256, 0, stream>>>(W_ih0, wih16, (long)2 * G4 * DIM);

  const size_t shmem = (size_t)ROWS_WG * HDIM * 2 + HDIM * 4 + 64 * 4;

  // ---- layer 0 ----
  gemm_pre_kernel<<<dim3(32, 32, 2), 256, 0, stream>>>(
      x16, wih16, b_ih, b_hh, pre, DIM);
  lstm_recur_kernel<<<2 * NWG_DIR, 256, shmem, stream>>>(
      pre, whh16, hf0, hb0, cfin, counters);

  // ---- layer 1 ----
  build_x1_kernel<<<(T_LEN * 2048) / 256, 256, 0, stream>>>(hf0, hb0, x16);
  f32_to_f16_kernel<<<8192, 256, 0, stream>>>(W_ih1, wih16, (long)2 * G4 * 2048);
  gemm_pre_kernel<<<dim3(32, 32, 2), 256, 0, stream>>>(
      x16, wih16, b_ih + 2 * G4, b_hh + 2 * G4, pre, 2048);
  lstm_recur_kernel<<<2 * NWG_DIR, 256, shmem, stream>>>(
      pre, whh16 + (size_t)2 * G4 * HDIM, hf1, hb1, cfin + 2 * HDIM, counters + 2);

  // ---- outputs ----
  float* out_t = (float*)d_out;                       // [T, 2048]
  float* enc   = out_t + (size_t)T_LEN * 2048;        // [510, 2048]
  float* hn    = enc + (size_t)(NSEG - 2) * 2048;     // [4, 1, 1024]
  float* cn    = hn + 4 * HDIM;                       // [4, 1, 1024]

  finalize_out_kernel<<<(T_LEN * 2048) / 256, 256, 0, stream>>>(hf1, hb1, out_t);
  finalize_states_kernel<<<32, 256, 0, stream>>>(hf0, hb0, hf1, hb1, cfin, hn, cn);
  segpool_kernel<<<NSEG, 256, 0, stream>>>(out_t, segment_ids, enc);
}
